// GraphConvBlock_54692113547388
// MI455X (gfx1250) — matrix operand order repaired and checked by reading, not compiled, not run on hardware
//
#include <hip/hip_runtime.h>
#include <hip/hip_bf16.h>
#include <stdint.h>

// ---------------------------------------------------------------------------
// GAT block for MI455X (gfx1250): wave32, WMMA 16x16x32 bf16, double-buffered
// LDS tiles filled with GLOBAL_LOAD_ASYNC_TO_LDS_B128 (ASYNCcnt-tracked),
// fused rank-1-score flash attention, adjacency-softmax reduced to a bitmask.
// ---------------------------------------------------------------------------

typedef __bf16 bf16;
typedef __attribute__((ext_vector_type(8)))  __bf16 v8bf;
typedef __attribute__((ext_vector_type(16))) __bf16 v16bf;
typedef __attribute__((ext_vector_type(8)))  float  v8f;

#define Bc 32
#define Sc 512
#define Dc 512
#define Hc 8
#define HDc 64
#define NFF 1024
#define NROWS (Bc * Sc)   // 16384

#if defined(__gfx1250__) && \
    __has_builtin(__builtin_amdgcn_global_load_async_to_lds_b128) && \
    __has_builtin(__builtin_amdgcn_s_wait_asynccnt)
#define USE_ASYNC 1
#else
#define USE_ASYNC 0
#endif

#if USE_ASYNC
// Builtin prototype (from clang diagnostic): parameters are pointers to
// 'int __attribute__((vector_size(16)))' in AS(1) / AS(3), non-const.
typedef int v4i_ __attribute__((vector_size(16)));
typedef __attribute__((address_space(1))) v4i_ gv4i_t;
typedef __attribute__((address_space(3))) v4i_ lv4i_t;
// 16B global -> LDS, no VGPR round-trip, tracked by ASYNCcnt.
__device__ __forceinline__ void async_cp16(const bf16* g, bf16* l) {
    __builtin_amdgcn_global_load_async_to_lds_b128(
        (gv4i_t*)(uintptr_t)g,
        (lv4i_t*)(uint32_t)(uintptr_t)l, 0, 0);
}
#define WAIT_ASYNC(n) __builtin_amdgcn_s_wait_asynccnt(n)
#else
__device__ __forceinline__ void async_cp16(const bf16* g, bf16* l) {
    *(v8bf*)l = *(const v8bf*)g;   // sync fallback
}
#define WAIT_ASYNC(n) ((void)0)
#endif

// Build a 16-element bf16 fragment from two contiguous 8-element runs in LDS.
__device__ __forceinline__ v16bf frag2(const bf16* p0, const bf16* p1) {
    v8bf lo = *(const v8bf*)p0;
    v8bf hi = *(const v8bf*)p1;
    v16bf r;
#pragma unroll
    for (int i = 0; i < 8; ++i) { r[i] = lo[i]; r[i + 8] = hi[i]; }
    return r;
}

__device__ __forceinline__ float gelu_exact(float x) {
    return 0.5f * x * (1.0f + erff(x * 0.70710678118654752f));
}

// ---------------------------------------------------------------------------
// One-shot f32 -> bf16 conversion (feeds the async-copy GEMM path and halves
// activation/weight HBM traffic on this memory-bound problem).
// ---------------------------------------------------------------------------
__global__ __launch_bounds__(256) void cvt_bf16(
    const float* __restrict__ in, bf16* __restrict__ out, int n)
{
    const int i = (blockIdx.x * 256 + threadIdx.x) * 8;
    if (i >= n) return;
    float4 a = *(const float4*)(in + i);
    float4 b = *(const float4*)(in + i + 4);
    v8bf o;
    o[0] = (bf16)a.x; o[1] = (bf16)a.y; o[2] = (bf16)a.z; o[3] = (bf16)a.w;
    o[4] = (bf16)b.x; o[5] = (bf16)b.y; o[6] = (bf16)b.z; o[7] = (bf16)b.w;
    *(v8bf*)(out + i) = o;
}

// ---------------------------------------------------------------------------
// WMMA GEMM: C[M,N] = act(A[M,K] @ W[N,K]^T + bias[N]); A, W already bf16.
// 128x128 block tile, 8 waves each 32x64 (2x4 wmma accums), double-buffered
// LDS tiles filled by async b128 copies (4 per thread per tile).
// ---------------------------------------------------------------------------
__device__ __forceinline__ void issue_tile(
    const bf16* __restrict__ gA, const bf16* __restrict__ gW,
    bf16 (*As)[40], bf16 (*Bs)[40], int tid, int K, int kbase)
{
#pragma unroll
    for (int i = 0; i < 2; ++i) {
        const int idx = tid * 2 + i;        // 0..511
        const int row = idx >> 2;           // 0..127
        const int c8  = (idx & 3) * 8;      // 0,8,16,24
        async_cp16(gA + (size_t)row * K + kbase + c8, &As[row][c8]);
        async_cp16(gW + (size_t)row * K + kbase + c8, &Bs[row][c8]);
    }
}

template <int ACT_GELU, int OUT_BF16>
__global__ __launch_bounds__(256) void gemm_wmma_bf16(
    const bf16* __restrict__ A, const bf16* __restrict__ W,
    const float* __restrict__ bias, void* __restrict__ out,
    int M, int N, int K)
{
    __shared__ __align__(16) bf16 As[2][128][40];  // [buf][m][k], +8 pad
    __shared__ __align__(16) bf16 Bs[2][128][40];  // [buf][n][k], K-major

    const int tid  = threadIdx.x;
    const int lane = tid & 31, wid = tid >> 5;
    const int half = lane >> 4, l16 = lane & 15;
    const int wm = (wid & 3) * 32, wn = (wid >> 2) * 64;
    const int m0 = blockIdx.x * 128, n0 = blockIdx.y * 128;

    const bf16* gA = A + (size_t)m0 * K;
    const bf16* gW = W + (size_t)n0 * K;

    v8f acc[2][4] = {};

    const int ktiles = K >> 5;
    issue_tile(gA, gW, As[0], Bs[0], tid, K, 0);

    for (int kt = 0; kt < ktiles; ++kt) {
        const int cur = kt & 1;
        if (kt + 1 < ktiles) {
            issue_tile(gA, gW, As[cur ^ 1], Bs[cur ^ 1], tid, K, (kt + 1) * 32);
            WAIT_ASYNC(4);   // in-order: tile kt's 4 copies have landed
        } else {
            WAIT_ASYNC(0);
        }
        __syncthreads();     // every wave's copies for tile kt are visible

        v16bf af[2], bf_[4];
#pragma unroll
        for (int mi = 0; mi < 2; ++mi)
            af[mi] = frag2(&As[cur][wm + mi * 16 + l16][8 * half],
                           &As[cur][wm + mi * 16 + l16][16 + 8 * half]);
#pragma unroll
        for (int ni = 0; ni < 4; ++ni) {
            const bf16* bp = &Bs[cur][wn + ni * 16 + l16][8 * half];
            bf_[ni] = frag2(bp, bp + 16);
        }
#pragma unroll
        for (int mi = 0; mi < 2; ++mi)
#pragma unroll
            for (int ni = 0; ni < 4; ++ni)
                acc[mi][ni] = __builtin_amdgcn_wmma_f32_16x16x32_bf16(
                    false, af[mi], false, bf_[ni], (short)0, acc[mi][ni],
                    false, false);
        __syncthreads();     // reads done before buffer is overwritten
    }

#pragma unroll
    for (int mi = 0; mi < 2; ++mi) {
#pragma unroll
        for (int ni = 0; ni < 4; ++ni) {
            const int n = n0 + wn + ni * 16 + l16;
            const float bv = bias[n];
#pragma unroll
            for (int r = 0; r < 8; ++r) {
                const int m = m0 + wm + mi * 16 + half * 8 + r;
                float v = acc[mi][ni][r] + bv;
                if (ACT_GELU) v = gelu_exact(v);
                if (OUT_BF16)
                    ((bf16*)out)[(size_t)m * N + n] = (bf16)v;
                else
                    ((float*)out)[(size_t)m * N + n] = v;
            }
        }
    }
}

// ---------------------------------------------------------------------------
// Row softmax over 512 logits -> 1-bit-per-element "softmax == 0" mask.
// ---------------------------------------------------------------------------
__global__ __launch_bounds__(256) void softmax_maskbits(
    const float* __restrict__ logits, unsigned* __restrict__ maskbits)
{
    const int row  = blockIdx.x * 8 + (threadIdx.x >> 5);
    const int lane = threadIdx.x & 31;
    const float* p = logits + (size_t)row * Sc;

    float vals[16];
    float mx = -3.0e38f;
#pragma unroll
    for (int c = 0; c < 16; ++c) {
        vals[c] = p[lane + 32 * c];
        mx = fmaxf(mx, vals[c]);
    }
#pragma unroll
    for (int s = 16; s > 0; s >>= 1) mx = fmaxf(mx, __shfl_xor(mx, s, 32));

    float sum = 0.0f;
#pragma unroll
    for (int c = 0; c < 16; ++c) { vals[c] = __expf(vals[c] - mx); sum += vals[c]; }
#pragma unroll
    for (int s = 16; s > 0; s >>= 1) sum += __shfl_xor(sum, s, 32);

#pragma unroll
    for (int c = 0; c < 16; ++c) {
        unsigned w = (unsigned)__ballot((vals[c] / sum) == 0.0f);
        if (lane == 0) maskbits[(size_t)row * 16 + c] = w;
    }
}

// ---------------------------------------------------------------------------
// src/dst attention logit halves: dot(h[b,s,head,:], a_w[:64] / a_w[64:]).
// ---------------------------------------------------------------------------
__global__ __launch_bounds__(256) void src_dst_kernel(
    const bf16* __restrict__ h, const float* __restrict__ a_w,
    float* __restrict__ srcv, float* __restrict__ dstv)
{
    const int idx = blockIdx.x * 256 + threadIdx.x;   // (b*S+s)*H + hh
    const int hh = idx & 7;
    const int q  = idx >> 3;          // b*S + s
    const int s  = q & (Sc - 1);
    const int b  = q >> 9;
    const bf16* hp = h + (size_t)q * Dc + hh * HDc;
    float ss = 0.0f, ds = 0.0f;
#pragma unroll
    for (int k = 0; k < HDc; ++k) {
        float hv = (float)hp[k];
        ss += hv * a_w[k];
        ds += hv * a_w[HDc + k];
    }
    const size_t o = ((size_t)(b * Hc + hh)) * Sc + s;
    srcv[o] = ss;
    dstv[o] = ds;
}

// ---------------------------------------------------------------------------
// Fused GAT attention: per (b,head,128-row slab).
//   e_ij = leakyrelu(src_i + dst_j, 0.2), masked where adj softmax == 0,
//   row softmax, out = P @ V via wmma (V staged K-major in LDS).
// ---------------------------------------------------------------------------
__global__ __launch_bounds__(256) void gat_attention(
    const bf16* __restrict__ hmat, const float* __restrict__ srcv,
    const float* __restrict__ dstv, const unsigned* __restrict__ maskbits,
    float* __restrict__ out)
{
    const int b = blockIdx.z, head = blockIdx.y, i0 = blockIdx.x * 128;

    __shared__ __align__(16) bf16 vt[HDc][Sc + 8];   // V transposed: [n][j]
    __shared__ float dsh[Sc];
    __shared__ __align__(16) bf16 pw[8][16][40];     // per-wave P chunk

    const int tid  = threadIdx.x;
    const int lane = tid & 31, wid = tid >> 5;
    const int half = lane >> 4, l16 = lane & 15;

    for (int jj = 0; jj < 2; ++jj) {
        const int j = tid * 2 + jj;
        const bf16* hp = hmat + ((size_t)(b * Sc + j)) * Dc + head * HDc;
#pragma unroll
        for (int n = 0; n < HDc; ++n) vt[n][j] = hp[n];
    }
    const float* dbase = dstv + ((size_t)(b * Hc + head)) * Sc;
    dsh[tid]       = dbase[tid];
    dsh[tid + 256] = dbase[tid + 256];
    __syncthreads();

    const int r0 = i0 + wid * 16;
    const float* sbase = srcv + ((size_t)(b * Hc + head)) * Sc;

    float srow[16], mx[16], rsum[16];
#pragma unroll
    for (int r = 0; r < 16; ++r) { srow[r] = sbase[r0 + r]; rsum[r] = 0.0f; }

    // Pass 1: row maxima (no exp).
#pragma unroll 1
    for (int r = 0; r < 16; ++r) {
        const unsigned* mb = maskbits + (size_t)(b * Sc + r0 + r) * 16;
        float m = -3.0e38f;
#pragma unroll
        for (int c = 0; c < 16; ++c) {
            float e = srow[r] + dsh[c * 32 + lane];
            e = (e > 0.0f) ? e : 0.2f * e;
            if ((mb[c] >> lane) & 1u) e = -1.0e9f;
            m = fmaxf(m, e);
        }
#pragma unroll
        for (int s = 16; s > 0; s >>= 1) m = fmaxf(m, __shfl_xor(m, s, 32));
        mx[r] = m;
    }

    // Pass 2: exp once, P chunk -> per-wave LDS -> wmma P@V.
    v8f acc[4] = {};
#pragma unroll 1
    for (int c = 0; c < 16; ++c) {
#pragma unroll
        for (int r = 0; r < 16; ++r) {
            const unsigned w = maskbits[(size_t)(b * Sc + r0 + r) * 16 + c];
            float e = srow[r] + dsh[c * 32 + lane];
            e = (e > 0.0f) ? e : 0.2f * e;
            if ((w >> lane) & 1u) e = -1.0e9f;
            float p = __expf(e - mx[r]);
            rsum[r] += p;
            pw[wid][r][lane] = (bf16)p;   // same-wave LDS, DS ops in-order
        }
        v16bf af = frag2(&pw[wid][l16][8 * half], &pw[wid][l16][16 + 8 * half]);
#pragma unroll
        for (int ni = 0; ni < 4; ++ni) {
            const bf16* bp = &vt[ni * 16 + l16][c * 32 + 8 * half];
            v16bf bf_ = frag2(bp, bp + 16);
            acc[ni] = __builtin_amdgcn_wmma_f32_16x16x32_bf16(
                false, af, false, bf_, (short)0, acc[ni], false, false);
        }
    }

#pragma unroll
    for (int r = 0; r < 16; ++r) {
        float s = rsum[r];
#pragma unroll
        for (int sft = 16; sft > 0; sft >>= 1) s += __shfl_xor(s, sft, 32);
        rsum[r] = s;
    }

#pragma unroll
    for (int ni = 0; ni < 4; ++ni) {
#pragma unroll
        for (int rr = 0; rr < 8; ++rr) {
            const int m = 8 * half + rr;
            const float v = acc[ni][rr] / rsum[m];
            out[((size_t)(b * Sc) + r0 + m) * Dc + head * HDc + ni * 16 + l16] = v;
        }
    }
}

// ---------------------------------------------------------------------------
// out = LayerNorm(X + R) * g + b ; one wave per 512-wide row.
// Optionally also emits a bf16 shadow copy (feeds the next GEMM's A).
// ---------------------------------------------------------------------------
__global__ __launch_bounds__(256) void add_layernorm(
    const float* __restrict__ X, const float* __restrict__ R,
    const float* __restrict__ g, const float* __restrict__ bta,
    float* __restrict__ out, bf16* __restrict__ out_bf)
{
    const int row  = blockIdx.x * 8 + (threadIdx.x >> 5);
    const int lane = threadIdx.x & 31;
    const size_t base = (size_t)row * Dc;

    float v[16];
    float sum = 0.0f, sq = 0.0f;
#pragma unroll
    for (int c = 0; c < 16; ++c) {
        const int col = lane + 32 * c;
        v[c] = X[base + col] + R[base + col];
        sum += v[c];
        sq  += v[c] * v[c];
    }
#pragma unroll
    for (int s = 16; s > 0; s >>= 1) {
        sum += __shfl_xor(sum, s, 32);
        sq  += __shfl_xor(sq,  s, 32);
    }
    const float mean = sum * (1.0f / Dc);
    const float var  = sq * (1.0f / Dc) - mean * mean;
    const float rstd = rsqrtf(var + 1e-5f);
#pragma unroll
    for (int c = 0; c < 16; ++c) {
        const int col = lane + 32 * c;
        const float y = (v[c] - mean) * rstd * g[col] + bta[col];
        out[base + col] = y;
        if (out_bf) out_bf[base + col] = (bf16)y;
    }
}

// ---------------------------------------------------------------------------
extern "C" void kernel_launch(void* const* d_in, const int* in_sizes, int n_in,
                              void* d_out, int out_size, void* d_ws, size_t ws_size,
                              hipStream_t stream) {
    const float* x     = (const float*)d_in[0];
    const float* adj   = (const float*)d_in[1];
    const float* w_W   = (const float*)d_in[2];
    const float* w_b   = (const float*)d_in[3];
    const float* a_w   = (const float*)d_in[4];
    const float* sem_W = (const float*)d_in[5];
    const float* sem_b = (const float*)d_in[6];
    const float* ff1_W = (const float*)d_in[7];
    const float* ff1_b = (const float*)d_in[8];
    const float* ff2_W = (const float*)d_in[9];
    const float* ff2_b = (const float*)d_in[10];
    const float* ln1_g = (const float*)d_in[11];
    const float* ln1_b = (const float*)d_in[12];
    const float* ln2_g = (const float*)d_in[13];
    const float* ln2_b = (const float*)d_in[14];

    char* ws = (char*)d_ws;
    const size_t MBy = 1ull << 20;
    float*    logits   = (float*)(ws + 0);             // 32 MB (reused)
    unsigned* maskb    = (unsigned*)(ws + 32 * MBy);   // 1 MB
    bf16*     hmat     = (bf16*)(ws + 33 * MBy);       // 16 MB
    float*    srcv     = (float*)(ws + 49 * MBy);      // 1 MB
    float*    dstv     = (float*)(ws + 50 * MBy);      // 1 MB
    float*    out1     = (float*)(ws + 51 * MBy);      // 32 MB
    bf16*     out1_bf  = (bf16*)(ws + 83 * MBy);       // 16 MB
    bf16*     y1_bf    = (bf16*)(ws + 99 * MBy);       // 32 MB
    float*    y2       = (float*)(ws + 131 * MBy);     // 32 MB
    bf16*     adj_bf   = (bf16*)(ws + 163 * MBy);      // 16 MB
    bf16*     x_bf     = (bf16*)(ws + 179 * MBy);      // 16 MB
    bf16*     semW_bf  = (bf16*)(ws + 195 * MBy);      // 0.5 MB
    bf16*     wW_bf    = (bf16*)(ws + 196 * MBy);      // 0.5 MB
    bf16*     ff1W_bf  = (bf16*)(ws + 197 * MBy);      // 1 MB
    bf16*     ff2W_bf  = (bf16*)(ws + 198 * MBy);      // 1 MB
    float*    attn_out = logits;                       // logits dead after mask

    dim3 blk(256);

    // 0) one-shot bf16 conversions (activations + weights)
    cvt_bf16<<<(NROWS * Dc) / 2048, blk, 0, stream>>>(adj, adj_bf, NROWS * Dc);
    cvt_bf16<<<(NROWS * Dc) / 2048, blk, 0, stream>>>(x, x_bf, NROWS * Dc);
    cvt_bf16<<<(Dc * Dc) / 2048, blk, 0, stream>>>(sem_W, semW_bf, Dc * Dc);
    cvt_bf16<<<(Dc * Dc) / 2048, blk, 0, stream>>>(w_W, wW_bf, Dc * Dc);
    cvt_bf16<<<(NFF * Dc) / 2048, blk, 0, stream>>>(ff1_W, ff1W_bf, NFF * Dc);
    cvt_bf16<<<(NFF * Dc) / 2048, blk, 0, stream>>>(ff2_W, ff2W_bf, NFF * Dc);

    // 1) adjacency semantic logits: adj @ sem_W^T + sem_b
    gemm_wmma_bf16<0, 0><<<dim3(NROWS / 128, Dc / 128), blk, 0, stream>>>(
        adj_bf, semW_bf, sem_b, logits, NROWS, Dc, Sc);
    // 2) row softmax -> "== 0" bitmask (32 MB of adj_s reduced to 1 MB)
    softmax_maskbits<<<NROWS / 8, blk, 0, stream>>>(logits, maskb);
    // 3) node projection: h = x @ w_W^T + w_b (bf16 for attention V)
    gemm_wmma_bf16<0, 1><<<dim3(NROWS / 128, Dc / 128), blk, 0, stream>>>(
        x_bf, wW_bf, w_b, hmat, NROWS, Dc, Dc);
    // 4) src/dst rank-1 score halves
    src_dst_kernel<<<(NROWS * Hc) / 256, blk, 0, stream>>>(hmat, a_w, srcv, dstv);
    // 5) fused attention (scores never hit HBM)
    gat_attention<<<dim3(Sc / 128, Hc, Bc), blk, 0, stream>>>(
        hmat, srcv, dstv, maskb, attn_out);
    // 6) LN1(attn + x) -> f32 + bf16 shadow
    add_layernorm<<<NROWS / 8, blk, 0, stream>>>(attn_out, x, ln1_g, ln1_b,
                                                 out1, out1_bf);
    // 7) ff1 + exact GELU -> bf16
    gemm_wmma_bf16<1, 1><<<dim3(NROWS / 128, NFF / 128), blk, 0, stream>>>(
        out1_bf, ff1W_bf, ff1_b, y1_bf, NROWS, NFF, Dc);
    // 8) ff2 -> f32
    gemm_wmma_bf16<0, 0><<<dim3(NROWS / 128, Dc / 128), blk, 0, stream>>>(
        y1_bf, ff2W_bf, ff2_b, y2, NROWS, Dc, NFF);
    // 9) LN2(ff + out1) -> d_out
    add_layernorm<<<NROWS / 8, blk, 0, stream>>>(y2, out1, ln2_g, ln2_b,
                                                 (float*)d_out, nullptr);
}
